// Encoder_82411832475895
// MI455X (gfx1250) — compile-verified, matrix operand order checked by reading, not run on hardware
//
#include <hip/hip_runtime.h>
#include <hip/hip_bf16.h>
#include <math.h>

// ---------------- types / constants ----------------
typedef __attribute__((ext_vector_type(16))) __bf16 v16bf;
typedef __attribute__((ext_vector_type(8)))  __bf16 v8bf;
typedef __attribute__((ext_vector_type(8)))  float  v8f;

#define B_   4096
#define X_   512
#define H_   1024
#define Z_   128
#define L_   16
#define STD_ 1.0025031276057952f   // exp(0.5*0.005)

#define SROW 40                    // LDS row stride in bf16 (80B: 16B-aligned, bank-permuting)

// ---------------- WMMA helpers ----------------
// 16-bit A/B fragment layout (CDNA5 ISA 7.12.2): lane L holds row (L&15);
// lanes<16: K = {k0..k0+7, k0+16..k0+23}; lanes>=16: K = {k0+8..k0+15, k0+24..k0+31}.
__device__ __forceinline__ v16bf ld_frag(const __bf16* __restrict__ base,
                                         int row, int stride, int k0, int lane) {
  const __bf16* p = base + (size_t)row * stride + k0 + ((lane & 16) ? 8 : 0);
  v8bf lo = *(const v8bf*)(p);
  v8bf hi = *(const v8bf*)(p + 16);
  v16bf f;
#pragma unroll
  for (int i = 0; i < 8; ++i) { f[i] = lo[i]; f[i + 8] = hi[i]; }
  return f;
}

// fragment from an LDS-staged chunk (rows of 32 bf16, stride SROW) -> ds_load_b128 x2
__device__ __forceinline__ v16bf ld_frag_lds(const __bf16* sbuf, int row, int lane) {
  const __bf16* p = sbuf + row * SROW + ((lane & 16) ? 8 : 0);
  v8bf lo = *(const v8bf*)(p);
  v8bf hi = *(const v8bf*)(p + 16);
  v16bf f;
#pragma unroll
  for (int i = 0; i < 8; ++i) { f[i] = lo[i]; f[i + 8] = hi[i]; }
  return f;
}

__device__ __forceinline__ v8f wmma_bf16(v16bf a, v16bf b, v8f c) {
  return __builtin_amdgcn_wmma_f32_16x16x32_bf16(false, a, false, b, (short)0, c,
                                                 false, false);
}

__device__ __forceinline__ v8f zero8() {
  v8f z;
#pragma unroll
  for (int i = 0; i < 8; ++i) z[i] = 0.0f;
  return z;
}

__device__ __forceinline__ float sigmoidf_(float x) { return 1.0f / (1.0f + __expf(-x)); }

// async copy of one 16B segment: global -> LDS (ASYNCcnt-tracked DMA path)
__device__ __forceinline__ void async_b128(const __bf16* gp, __bf16* lp) {
  uint32_t lds_addr = (uint32_t)(uintptr_t)lp;           // low 32 bits = LDS offset
  asm volatile("global_load_async_to_lds_b128 %0, %1, off"
               :: "v"(lds_addr), "v"((uint64_t)(uintptr_t)gp)
               : "memory");
}
__device__ __forceinline__ void wait_async0() {
  asm volatile("s_wait_asynccnt 0x0" ::: "memory");
}

// ---------------- conversion kernels ----------------
__global__ void k_f32_to_bf16(const float* __restrict__ src, __bf16* __restrict__ dst,
                              int n) {
  int i = blockIdx.x * blockDim.x + threadIdx.x;
  int stride = gridDim.x * blockDim.x;
  for (; i < n; i += stride) dst[i] = (__bf16)src[i];
}

// Split gru_Wih [rows,513] -> bf16 [rows,512] + f32 last column
__global__ void k_split_wih(const float* __restrict__ src, __bf16* __restrict__ wx,
                            float* __restrict__ wcol, int rows) {
  int i = blockIdx.x * blockDim.x + threadIdx.x;
  int stride = gridDim.x * blockDim.x;
  int n = rows * 513;
  for (; i < n; i += stride) {
    int r = i / 513, c = i - r * 513;
    float v = src[i];
    if (c < 512) wx[(size_t)r * 512 + c] = (__bf16)v;
    else         wcol[r] = v;
  }
}

// ---------------- predict head: pbuf[row, nblk] = sum_n relu(A@W1.T+b1)*W2 ----------------
__global__ __launch_bounds__(128)
void k_predict(const __bf16* __restrict__ A, int K,
               const __bf16* __restrict__ W1, const float* __restrict__ b1,
               const float* __restrict__ W2, float* __restrict__ pbuf) {
  __shared__ __bf16 sB[2][64 * SROW];          // 64 rows x 32 bf16 (+pad), double-buffered
  const int tid  = threadIdx.x;
  const int lane = tid & 31;
  const int wave = tid >> 5;
  const int m_base = blockIdx.x * 64 + wave * 16;
  const int n_base = blockIdx.y * 64;
  const int rowA = m_base + (lane & 15);
  const int nch = K >> 5;

  v8f acc[4];
#pragma unroll
  for (int t = 0; t < 4; ++t) acc[t] = zero8();

  auto stage = [&](int c, int buf) {           // 64 rows x 4 segs = 256 segs / 128 thr
#pragma unroll
    for (int j = 0; j < 2; ++j) {
      int s = tid * 2 + j;
      int row = s >> 2, seg = s & 3;
      const __bf16* gp = W1 + (size_t)(n_base + row) * K + (c << 5) + seg * 8;
      async_b128(gp, &sB[buf][row * SROW + seg * 8]);
    }
  };

  stage(0, 0);
  for (int c = 0; c < nch; ++c) {
    int cur = c & 1;
    wait_async0();
    __syncthreads();
    if (c + 1 < nch) stage(c + 1, 1 - cur);
    // hoist all fragment loads, then burst the WMMAs (batched dscnt wait)
    v16bf a = ld_frag(A, rowA, K, c << 5, lane);
    v16bf bf[4];
#pragma unroll
    for (int t = 0; t < 4; ++t)
      bf[t] = ld_frag_lds(&sB[cur][0], t * 16 + (lane & 15), lane);
#pragma unroll
    for (int t = 0; t < 4; ++t)
      acc[t] = wmma_bf16(a, bf[t], acc[t]);
  }

  float part[8];
#pragma unroll
  for (int v = 0; v < 8; ++v) part[v] = 0.0f;
#pragma unroll
  for (int t = 0; t < 4; ++t) {
    int col = n_base + t * 16 + (lane & 15);
    float b1v = b1[col], w2v = W2[col];
#pragma unroll
    for (int v = 0; v < 8; ++v) {
      float h = acc[t][v] + b1v;
      h = h > 0.0f ? h : 0.0f;
      part[v] += h * w2v;
    }
  }
#pragma unroll
  for (int v = 0; v < 8; ++v) {
#pragma unroll
    for (int m = 1; m < 16; m <<= 1) part[v] += __shfl_xor(part[v], m, 32);
  }
  if ((lane & 15) == 0) {
    int rofs = (lane >> 4) << 3;
#pragma unroll
    for (int v = 0; v < 8; ++v)
      pbuf[(size_t)(m_base + rofs + v) * 16 + blockIdx.y] = part[v];
  }
}

// deterministic reduction of the 16 N-block partials -> t pre-activation
__global__ void k_reduce_t(const float* __restrict__ pbuf, float* __restrict__ tpre) {
  int row = blockIdx.x * blockDim.x + threadIdx.x;
  if (row >= B_) return;
  float s = 0.0f;
#pragma unroll
  for (int j = 0; j < 16; ++j) s += pbuf[(size_t)row * 16 + j];
  tpre[row] = s;
}

// ---------------- fused GRU step: gates GEMMs + gate math + reparam ----------------
__global__ __launch_bounds__(128)
void k_gates(const __bf16* __restrict__ xbf, const __bf16* __restrict__ zbf,
             const float* __restrict__ zf,
             const __bf16* __restrict__ Wx, const float* __restrict__ wcol,
             const __bf16* __restrict__ Whh,
             const float* __restrict__ bih, const float* __restrict__ bhh,
             const float* __restrict__ tpre, const float* __restrict__ b2,
             const float* __restrict__ eps,
             float* __restrict__ zf_out, __bf16* __restrict__ zbf_out,
             float* __restrict__ tout, int l, int first) {
  __shared__ __bf16 sB[2][96 * SROW];          // 3 gate blocks x 32 rows, double-buffered
  const int tid  = threadIdx.x;
  const int lane = tid & 31;
  const int wave = tid >> 5;
  const int m_base = blockIdx.x * 64 + wave * 16;
  const int n_base = blockIdx.y * 32;
  const int rowA = m_base + (lane & 15);
  const float b2v = b2[0];
  const int nch = first ? 16 : 48;             // 16 x-chunks (+32 z-chunks)

  v8f aR[2], aU[2], aNx[2], aNh[2];
#pragma unroll
  for (int t = 0; t < 2; ++t) { aR[t] = zero8(); aU[t] = zero8(); aNx[t] = zero8(); aNh[t] = zero8(); }

  auto stage = [&](int c, int buf) {           // 96 rows x 4 segs = 384 segs / 128 thr
    const __bf16* Wsrc = (c < 16) ? Wx : Whh;
    const int stride   = (c < 16) ? X_ : H_;
    const int k0       = ((c < 16) ? c : (c - 16)) << 5;
#pragma unroll
    for (int j = 0; j < 3; ++j) {
      int s = tid * 3 + j;
      int row96 = s >> 2, seg = s & 3;
      int g = row96 >> 5, r = row96 & 31;
      const __bf16* gp = Wsrc + (size_t)(g * H_ + n_base + r) * stride + k0 + seg * 8;
      async_b128(gp, &sB[buf][row96 * SROW + seg * 8]);
    }
  };

  stage(0, 0);
  for (int c = 0; c < nch; ++c) {
    int cur = c & 1;
    wait_async0();
    __syncthreads();
    if (c + 1 < nch) stage(c + 1, 1 - cur);

    const bool xpart = (c < 16);
    const __bf16* Abase = xpart ? xbf : zbf;
    const int astride   = xpart ? X_ : H_;
    const int k0        = (xpart ? c : (c - 16)) << 5;
    // hoist all fragment loads, then burst the 6 WMMAs (batched dscnt wait)
    v16bf a = ld_frag(Abase, rowA, astride, k0, lane);
    v16bf bf[6];
#pragma unroll
    for (int t = 0; t < 2; ++t) {
      int rn = t * 16 + (lane & 15);
      bf[t * 3 + 0] = ld_frag_lds(&sB[cur][0], 0 * 32 + rn, lane);
      bf[t * 3 + 1] = ld_frag_lds(&sB[cur][0], 1 * 32 + rn, lane);
      bf[t * 3 + 2] = ld_frag_lds(&sB[cur][0], 2 * 32 + rn, lane);
    }
#pragma unroll
    for (int t = 0; t < 2; ++t) {
      aR[t] = wmma_bf16(a, bf[t * 3 + 0], aR[t]);
      aU[t] = wmma_bf16(a, bf[t * 3 + 1], aU[t]);
      if (xpart) aNx[t] = wmma_bf16(a, bf[t * 3 + 2], aNx[t]);
      else       aNh[t] = wmma_bf16(a, bf[t * 3 + 2], aNh[t]);
    }
  }

  // epilogue: C/D layout -> row = m_base + v + 8*(lane/16), col = n_base + t*16 + (lane&15)
  const int rofs = (lane >> 4) << 3;
#pragma unroll
  for (int v = 0; v < 8; ++v) {
    int row = m_base + rofs + v;
    float tmp = (tpre[row] + b2v) >= 0.0f ? 1.0f : 0.0f;   // (sigmoid(t) >= 0.5)
#pragma unroll
    for (int t = 0; t < 2; ++t) {
      int col = n_base + t * 16 + (lane & 15);
      float r  = sigmoidf_(aR[t][v] + bih[col]      + bhh[col]      + tmp * wcol[col]);
      float u  = sigmoidf_(aU[t][v] + bih[H_ + col] + bhh[H_ + col] + tmp * wcol[H_ + col]);
      float gn = aNx[t][v] + bih[2 * H_ + col] + tmp * wcol[2 * H_ + col]
               + r * (aNh[t][v] + bhh[2 * H_ + col]);
      float nn = tanhf(gn);
      float zo = first ? 0.0f : zf[(size_t)row * H_ + col];
      float mu = (1.0f - u) * nn + u * zo;
      float zn = mu + STD_ * eps[(size_t)row * H_ + col];
      zf_out[(size_t)row * H_ + col]  = zn;
      zbf_out[(size_t)row * H_ + col] = (__bf16)zn;
    }
  }

  // emit t[:, l] = sigmoid(pre-activation) once (N-block 0 only)
  if (blockIdx.y == 0 && threadIdx.x < 64) {
    int row = blockIdx.x * 64 + threadIdx.x;
    tout[(size_t)row * L_ + l] = sigmoidf_(tpre[row] + b2v);
  }
}

// ---------------- final heads: mu / softplus(log_var) / latent ----------------
__global__ __launch_bounds__(128)
void k_final(const __bf16* __restrict__ zbf, const __bf16* __restrict__ muW,
             const __bf16* __restrict__ lvW, const float* __restrict__ mub,
             const float* __restrict__ lvb, const float* __restrict__ epsf,
             float* __restrict__ out) {
  const int lane = threadIdx.x & 31;
  const int wave = threadIdx.x >> 5;
  const int m_base = blockIdx.x * 64 + wave * 16;
  const int n_base = blockIdx.y * 16;
  const int rowA = m_base + (lane & 15);

  v8f am = zero8(), al = zero8();
  for (int k0 = 0; k0 < H_; k0 += 32) {
    v16bf a  = ld_frag(zbf, rowA, H_, k0, lane);
    v16bf bm = ld_frag(muW, n_base + (lane & 15), H_, k0, lane);
    v16bf bl = ld_frag(lvW, n_base + (lane & 15), H_, k0, lane);
    am = wmma_bf16(a, bm, am);
    al = wmma_bf16(a, bl, al);
  }
  const int rofs = (lane >> 4) << 3;
  int col = n_base + (lane & 15);
  float mb = mub[col], lb = lvb[col];
#pragma unroll
  for (int v = 0; v < 8; ++v) {
    int row = m_base + rofs + v;
    float m   = am[v] + mb;
    float lvp = al[v] + lb;
    float sp  = (lvp > 0.0f ? lvp : 0.0f) + log1pf(__expf(-fabsf(lvp)));  // softplus
    float lat = m + __expf(0.5f * sp) * epsf[(size_t)row * Z_ + col];
    size_t o = (size_t)row * Z_ + col;
    out[o]                       = lat;  // latent
    out[(size_t)B_ * Z_ + o]     = m;    // mu_f
    out[2 * (size_t)B_ * Z_ + o] = sp;   // log_var
  }
}

// ---------------- host-side launch sequence ----------------
extern "C" void kernel_launch(void* const* d_in, const int* in_sizes, int n_in,
                              void* d_out, int out_size, void* d_ws, size_t ws_size,
                              hipStream_t stream) {
  (void)in_sizes; (void)n_in; (void)out_size; (void)ws_size;
  const float* x     = (const float*)d_in[0];
  const float* fcW1  = (const float*)d_in[1];
  const float* fcb1  = (const float*)d_in[2];
  const float* fcW2  = (const float*)d_in[3];
  const float* fcb2  = (const float*)d_in[4];
  const float* gWih  = (const float*)d_in[5];
  const float* gWhh  = (const float*)d_in[6];
  const float* gbih  = (const float*)d_in[7];
  const float* gbhh  = (const float*)d_in[8];
  const float* pW1   = (const float*)d_in[9];
  const float* pb1   = (const float*)d_in[10];
  const float* pW2   = (const float*)d_in[11];
  const float* pb2   = (const float*)d_in[12];
  const float* muW   = (const float*)d_in[13];
  const float* mub   = (const float*)d_in[14];
  const float* lvW   = (const float*)d_in[15];
  const float* lvb   = (const float*)d_in[16];
  const float* eps_s = (const float*)d_in[17];
  const float* eps_f = (const float*)d_in[18];
  float* out = (float*)d_out;

  char* ws = (char*)d_ws;
  size_t off = 0;
  auto alloc = [&](size_t bytes) -> void* {
    void* p = ws + off;
    off = (off + bytes + 255) & ~(size_t)255;
    return p;
  };
  __bf16* xbf    = (__bf16*)alloc((size_t)B_ * X_ * 2);
  __bf16* fcW1bf = (__bf16*)alloc((size_t)H_ * X_ * 2);
  __bf16* Wxbf   = (__bf16*)alloc((size_t)L_ * 3 * H_ * X_ * 2);
  float*  wcol   = (float*) alloc((size_t)L_ * 3 * H_ * 4);
  __bf16* Whhbf  = (__bf16*)alloc((size_t)L_ * 3 * H_ * H_ * 2);
  __bf16* pW1bf  = (__bf16*)alloc((size_t)L_ * H_ * H_ * 2);
  __bf16* muWbf  = (__bf16*)alloc((size_t)Z_ * H_ * 2);
  __bf16* lvWbf  = (__bf16*)alloc((size_t)Z_ * H_ * 2);
  float*  zf0    = (float*) alloc((size_t)B_ * H_ * 4);
  float*  zf1    = (float*) alloc((size_t)B_ * H_ * 4);
  __bf16* zb0    = (__bf16*)alloc((size_t)B_ * H_ * 2);
  __bf16* zb1    = (__bf16*)alloc((size_t)B_ * H_ * 2);
  float*  tpre   = (float*) alloc((size_t)(L_ + 1) * B_ * 4);
  float*  pbuf   = (float*) alloc((size_t)B_ * 16 * 4);

  const int cb = 256;
  k_f32_to_bf16<<<2048, cb, 0, stream>>>(x,    xbf,    B_ * X_);
  k_f32_to_bf16<<<1024, cb, 0, stream>>>(fcW1, fcW1bf, H_ * X_);
  k_split_wih <<<8192, cb, 0, stream>>>(gWih, Wxbf, wcol, L_ * 3 * H_);
  k_f32_to_bf16<<<8192, cb, 0, stream>>>(gWhh, Whhbf, L_ * 3 * H_ * H_);
  k_f32_to_bf16<<<4096, cb, 0, stream>>>(pW1,  pW1bf, L_ * H_ * H_);
  k_f32_to_bf16<<<256,  cb, 0, stream>>>(muW,  muWbf, Z_ * H_);
  k_f32_to_bf16<<<256,  cb, 0, stream>>>(lvW,  lvWbf, Z_ * H_);

  // firstcell -> tpre[0]
  k_predict<<<dim3(B_ / 64, H_ / 64), 128, 0, stream>>>(xbf, X_, fcW1bf, fcb1, fcW2, pbuf);
  k_reduce_t<<<B_ / 256, 256, 0, stream>>>(pbuf, tpre);

  float*  zf_cur = zf0; float*  zf_nxt = zf1;
  __bf16* zb_cur = zb0; __bf16* zb_nxt = zb1;
  for (int l = 0; l < L_; ++l) {
    const float* b2 = (l == 0) ? fcb2 : (pb2 + (l - 1));
    k_gates<<<dim3(B_ / 64, H_ / 32), 128, 0, stream>>>(
        xbf, zb_cur, zf_cur,
        Wxbf + (size_t)l * 3 * H_ * X_, wcol + (size_t)l * 3 * H_,
        Whhbf + (size_t)l * 3 * H_ * H_,
        gbih + (size_t)l * 3 * H_, gbhh + (size_t)l * 3 * H_,
        tpre + (size_t)l * B_, b2,
        eps_s + (size_t)l * B_ * H_,
        zf_nxt, zb_nxt, out + 3 * (size_t)B_ * Z_, l, l == 0 ? 1 : 0);
    if (l < L_ - 1) {  // t_16 is never consumed by the outputs
      k_predict<<<dim3(B_ / 64, H_ / 64), 128, 0, stream>>>(
          zb_nxt, H_, pW1bf + (size_t)l * H_ * H_, pb1 + (size_t)l * H_,
          pW2 + (size_t)l * H_, pbuf);
      k_reduce_t<<<B_ / 256, 256, 0, stream>>>(pbuf, tpre + (size_t)(l + 1) * B_);
    }
    float*  tf = zf_cur; zf_cur = zf_nxt; zf_nxt = tf;
    __bf16* tb = zb_cur; zb_cur = zb_nxt; zb_nxt = tb;
  }

  k_final<<<dim3(B_ / 64, Z_ / 16), 128, 0, stream>>>(zb_cur, muWbf, lvWbf, mub, lvb,
                                                      eps_f, out);
}
